// BiLSTMLM_67336497267397
// MI455X (gfx1250) — matrix operand order repaired and checked by reading, not compiled, not run on hardware
//
#include <hip/hip_runtime.h>
#include <hip/hip_bf16.h>

// ---------------- problem constants ----------------
#define HID   16
#define EMBD  32
#define SEQ   64
#define BATCH 32
#define VOC   50257
#define VOCP  51200          // VOC rounded up to 50*1024 (multiple of 128)
#define NROW  (SEQ*BATCH)    // 2048 rows of the big GEMM
#define KDIM  32             // 2*HID, matches WMMA f16 K exactly

// workspace layout (bytes)
#define X_OFF    ((size_t)0)          // X: 2048*64 f32             = 524288
#define CH_OFF   ((size_t)524288)     // concat_h f16: 2048*32*2    = 131072
#define W16_OFF  ((size_t)655360)     // W_ho f16 padded: VOCP*32*2 = 3276800
#define BHO_OFF  ((size_t)3932160)    // padded bias: VOCP*4        = 204800
#define P_OFF    ((size_t)4136960)    // partials: 2048*7 f32       = 57344
#define N_OFF    ((size_t)4194304)    // norm: 2048 f32             = 8192

#define CW   8192                     // cols per stats chunk
#define NCH  7                        // 6 full chunks + 2048-col chunk (VOCP/CW)

typedef _Float16 half8  __attribute__((ext_vector_type(8)));
typedef _Float16 half16 __attribute__((ext_vector_type(16)));
typedef float    v8f    __attribute__((ext_vector_type(8)));

__device__ __forceinline__ float fsigmoid(float x){ return 1.0f/(1.0f+__expf(-x)); }
__device__ __forceinline__ float ftanh(float x){ return 1.0f - 2.0f/(__expf(2.0f*x)+1.0f); }

// ------------------------------------------------------------------
// 1) X[row][gh] = b_g[j] + emb[tok[row]] . W_g[j][0:32]  (no recurrence)
// ------------------------------------------------------------------
__global__ void k_prep_x(const int* __restrict__ tok, const float* __restrict__ emb,
    const float* __restrict__ Wf, const float* __restrict__ bf,
    const float* __restrict__ Wi, const float* __restrict__ bi,
    const float* __restrict__ Wc, const float* __restrict__ bc,
    const float* __restrict__ Wo, const float* __restrict__ bo,
    float* __restrict__ X) {
  const int row = blockIdx.x;
  const int t = threadIdx.x;          // 0..63 == gh
  const int g = t >> 4, j = t & 15;
  const float* W; const float* b;
  if (g==0){W=Wf;b=bf;} else if(g==1){W=Wi;b=bi;} else if(g==2){W=Wc;b=bc;} else {W=Wo;b=bo;}
  const float* e  = emb + (size_t)tok[row]*EMBD;
  const float* wr = W + j*(EMBD+HID);
  float acc = b[j];
  #pragma unroll
  for (int k=0;k<EMBD;k++) acc += e[k]*wr[k];
  X[row*64 + t] = acc;
}

// ------------------------------------------------------------------
// 2) padded f16 weight + padded bias (-inf tail => exp()=0, branchless GEMM)
// ------------------------------------------------------------------
__global__ void k_prep_w(const float* __restrict__ W, const float* __restrict__ bho,
                         _Float16* __restrict__ W16, float* __restrict__ bhoP){
  const int stride = gridDim.x*blockDim.x;
  for (int i = blockIdx.x*blockDim.x + threadIdx.x; i < VOCP*KDIM; i += stride)
    W16[i] = (i < VOC*KDIM) ? (_Float16)W[i] : (_Float16)0.f;
  for (int i = blockIdx.x*blockDim.x + threadIdx.x; i < VOCP; i += stride)
    bhoP[i] = (i < VOC) ? bho[i] : -INFINITY;
}

// ------------------------------------------------------------------
// 3) recurrent scans. blockIdx.x==0: forward, ==1: backward.
//    thread t owns cell (b,j); C in registers, h via LDS.
// ------------------------------------------------------------------
__global__ void __launch_bounds__(512) k_scan(const float* __restrict__ X,
    const float* __restrict__ h0, const float* __restrict__ C0,
    const float* __restrict__ Wf, const float* __restrict__ Wi,
    const float* __restrict__ Wc, const float* __restrict__ Wo,
    _Float16* __restrict__ CH) {
  __shared__ float WhT[16*64];        // [k][gh]
  __shared__ float hl[BATCH*HID];
  const int dir = blockIdx.x;
  const int t = threadIdx.x;
  const int b = t >> 4, j = t & 15;
  for (int idx = t; idx < 16*64; idx += 512) {
    int k = idx >> 6, gh = idx & 63, g = gh >> 4, jj = gh & 15;
    const float* W = (g==0)?Wf:(g==1)?Wi:(g==2)?Wc:Wo;
    WhT[idx] = W[jj*(EMBD+HID) + EMBD + k];
  }
  float C = C0[j];
  float h = h0[j];
  hl[t] = h;
  if (dir == 0) CH[(0*BATCH + b)*KDIM + j] = (_Float16)h;                 // lefts[0]
  else          CH[((SEQ-1)*BATCH + b)*KDIM + HID + j] = (_Float16)h;     // rights[S-1]
  __syncthreads();
  for (int step = 0; step < SEQ; ++step) {
    const int s = dir ? (SEQ-1-step) : step;
    const float* xr = X + ((size_t)(s*BATCH + b))*64;
    float pf = xr[j], pi = xr[16+j], pc = xr[32+j], po = xr[48+j];
    const float* hb = hl + b*HID;
    #pragma unroll
    for (int k=0;k<HID;k++){
      float hv = hb[k];
      const float* wk = WhT + k*64;
      pf += hv*wk[j]; pi += hv*wk[16+j]; pc += hv*wk[32+j]; po += hv*wk[48+j];
    }
    float f = fsigmoid(pf), ig = fsigmoid(pi), ct = ftanh(pc), o = fsigmoid(po);
    C = f*C + ig*ct;
    float hn = o*ftanh(C);
    __syncthreads();
    hl[t] = hn;
    __syncthreads();
    if (dir == 0) { if (s < SEQ-1) CH[((s+1)*BATCH + b)*KDIM + j] = (_Float16)hn; }
    else          { if (s > 0)     CH[((s-1)*BATCH + b)*KDIM + HID + j] = (_Float16)hn; }
  }
}

// ---- A-fragment loader: concat_h row, contiguous k = 16*hf + e (B loads use the same order) ----
__device__ __forceinline__ half16 load_A(const _Float16* __restrict__ CH, int row0, int lane){
  const int hf = lane >> 4, m = lane & 15;
  const _Float16* ar = CH + (size_t)(row0 + m)*KDIM;
  half8 a0 = *(const half8*)(ar + 16*hf);       // K = 16*hf + e      (e=0..7)
  half8 a1 = *(const half8*)(ar + 16*hf + 8);   // K = 16*hf + e      (e=8..15)
  return __builtin_shufflevector(a0,a1, 0,1,2,3,4,5,6,7,8,9,10,11,12,13,14,15);
}

// ------------------------------------------------------------------
// 4) stats: per-row sum(exp(logit)) over a column chunk via WMMA.
//    |logit| <= ~8.3 by construction => no max-shift needed.
//    Branchless + manual 2-tile pipeline (tile counts divisible by 8,
//    so no tail): second WMMA fills the first's WMMA->VALU hazard slots.
// ------------------------------------------------------------------
__global__ void __launch_bounds__(128) k_stats(const _Float16* __restrict__ CH,
    const _Float16* __restrict__ W16, const float* __restrict__ bhoP,
    float* __restrict__ part) {
  const int row0 = blockIdx.x * 16;
  const int base = blockIdx.y * CW;
  const int t = threadIdx.x;
  const int wave = t >> 5, lane = t & 31;
  const int hf = lane >> 4, n = lane & 15;
  const int nTiles = (min(CW, VOCP - base)) >> 4;     // 512 or 128: divisible by 8

  const half16 a = load_A(CH, row0, lane);

  float ls0[8], ls1[8];
  #pragma unroll
  for (int r=0;r<8;r++){ ls0[r] = 0.f; ls1[r] = 0.f; }

  for (int ti = wave; ti < nTiles; ti += 8) {
    const int col0 = base + ti*16 + n;
    const int col1 = col0 + 64;                       // tile ti+4
    const half16 b0 = *(const half16*)(W16 + (size_t)col0*KDIM + 16*hf);
    const half16 b1 = *(const half16*)(W16 + (size_t)col1*KDIM + 16*hf);
    const float bias0 = bhoP[col0];
    const float bias1 = bhoP[col1];
    v8f c0 = {}, c1 = {};
    c0 = __builtin_amdgcn_wmma_f32_16x16x32_f16(false, a, false, b0, (short)0, c0, false, false);
    c1 = __builtin_amdgcn_wmma_f32_16x16x32_f16(false, a, false, b1, (short)0, c1, false, false);
    #pragma unroll
    for (int r=0;r<8;r++) ls0[r] += __expf(c0[r] + bias0);
    #pragma unroll
    for (int r=0;r<8;r++) ls1[r] += __expf(c1[r] + bias1);
  }
  __shared__ float red[16][64];                 // [rowInTile][wave*16+n]
  #pragma unroll
  for (int r=0;r<8;r++) red[r + 8*hf][wave*16 + n] = ls0[r] + ls1[r];
  __syncthreads();
  if (t < 16) {
    float S = 0.f;
    #pragma unroll
    for (int i=0;i<64;i++) S += red[t][i];
    part[(size_t)(row0 + t)*NCH + blockIdx.y] = S;
  }
}

// ------------------------------------------------------------------
// 5) merge chunk partials -> norm[row] = log(sum exp(logits))
// ------------------------------------------------------------------
__global__ void k_merge(const float* __restrict__ part, float* __restrict__ norm) {
  const int row = blockIdx.x*blockDim.x + threadIdx.x;
  if (row >= NROW) return;
  float S = 0.f;
  #pragma unroll
  for (int c=0;c<NCH;c++) S += part[(size_t)row*NCH + c];
  norm[row] = logf(S);
}

// ---- guarded/unguarded tile store helper ----
__device__ __forceinline__ void store_tile(const v8f& c, float bias, const float* nrm,
                                           float* __restrict__ out, int row0, int hf,
                                           int col, int tileStart) {
  if (tileStart + 16 <= VOC) {                  // wave-uniform fast path
    #pragma unroll
    for (int r=0;r<8;r++)
      __builtin_nontemporal_store(c[r] + bias - nrm[r],
                                  out + (size_t)(row0 + r + 8*hf)*VOC + col);
  } else if (col < VOC) {                       // single boundary tile
    #pragma unroll
    for (int r=0;r<8;r++)
      __builtin_nontemporal_store(c[r] + bias - nrm[r],
                                  out + (size_t)(row0 + r + 8*hf)*VOC + col);
  }                                             // fully padded tile: no stores
}

// ------------------------------------------------------------------
// 6) write: redo WMMA GEMM, stream log_softmax to HBM with NT stores.
//    Unguarded loads (padded arrays), 2-tile pipeline, 64 tiles/block.
// ------------------------------------------------------------------
__global__ void __launch_bounds__(128) k_write(const _Float16* __restrict__ CH,
    const _Float16* __restrict__ W16, const float* __restrict__ bhoP,
    const float* __restrict__ norm, float* __restrict__ out) {
  const int row0 = blockIdx.x * 16;
  const int base = blockIdx.y * 1024;
  const int t = threadIdx.x;
  const int wave = t >> 5, lane = t & 31;
  const int hf = lane >> 4, n = lane & 15;

  const half16 a = load_A(CH, row0, lane);

  float nrm[8];
  #pragma unroll
  for (int r=0;r<8;r++) nrm[r] = norm[row0 + r + 8*hf];

  #pragma unroll 2
  for (int ti = wave; ti < 64; ti += 8) {       // 8 iters/wave, no tail
    const int s0 = base + ti*16;
    const int s1 = s0 + 64;                     // tile ti+4
    const int col0 = s0 + n;
    const int col1 = s1 + n;
    const half16 b0 = *(const half16*)(W16 + (size_t)col0*KDIM + 16*hf);
    const half16 b1 = *(const half16*)(W16 + (size_t)col1*KDIM + 16*hf);
    const float bias0 = bhoP[col0];
    const float bias1 = bhoP[col1];
    v8f c0 = {}, c1 = {};
    c0 = __builtin_amdgcn_wmma_f32_16x16x32_f16(false, a, false, b0, (short)0, c0, false, false);
    c1 = __builtin_amdgcn_wmma_f32_16x16x32_f16(false, a, false, b1, (short)0, c1, false, false);
    store_tile(c0, bias0, nrm, out, row0, hf, col0, s0);
    store_tile(c1, bias1, nrm, out, row0, hf, col1, s1);
  }
}

// ------------------------------------------------------------------
extern "C" void kernel_launch(void* const* d_in, const int* in_sizes, int n_in,
                              void* d_out, int out_size, void* d_ws, size_t ws_size,
                              hipStream_t stream) {
  const int*   tok = (const int*)  d_in[0];
  const float* emb = (const float*)d_in[1];
  const float* W_f = (const float*)d_in[2];  const float* b_f = (const float*)d_in[3];
  const float* W_i = (const float*)d_in[4];  const float* b_i = (const float*)d_in[5];
  const float* W_C = (const float*)d_in[6];  const float* b_C = (const float*)d_in[7];
  const float* W_o = (const float*)d_in[8];  const float* b_o = (const float*)d_in[9];
  const float* W_ho= (const float*)d_in[10]; const float* b_ho= (const float*)d_in[11];
  const float* h0  = (const float*)d_in[12]; const float* C0  = (const float*)d_in[13];
  float* out = (float*)d_out;
  char* ws = (char*)d_ws;

  float*     X    = (float*)    (ws + X_OFF);
  _Float16*  CH   = (_Float16*) (ws + CH_OFF);
  _Float16*  W16  = (_Float16*) (ws + W16_OFF);
  float*     bhoP = (float*)    (ws + BHO_OFF);
  float*     part = (float*)    (ws + P_OFF);
  float*     norm = (float*)    (ws + N_OFF);

  k_prep_x<<<NROW, 64, 0, stream>>>(tok, emb, W_f,b_f, W_i,b_i, W_C,b_C, W_o,b_o, X);
  k_prep_w<<<1600, 256, 0, stream>>>(W_ho, b_ho, W16, bhoP);
  k_scan<<<2, 512, 0, stream>>>(X, h0, C0, W_f, W_i, W_C, W_o, CH);
  k_stats<<<dim3(128, NCH), 128, 0, stream>>>(CH, W16, bhoP, part);
  k_merge<<<8, 256, 0, stream>>>(part, norm);
  k_write<<<dim3(128, 50), 128, 0, stream>>>(CH, W16, bhoP, norm, out);
}
